// Nonlocal_44590350467253
// MI455X (gfx1250) — matrix operand order repaired and checked by reading, not compiled, not run on hardware
//
#include <hip/hip_runtime.h>
#include <hip/hip_bf16.h>

typedef __attribute__((ext_vector_type(16))) _Float16 v16h;
typedef __attribute__((ext_vector_type(8)))  _Float16 v8h;
typedef __attribute__((ext_vector_type(8)))  float    v8f;

constexpr int NT   = 9216;   // h*w tokens
constexpr int CIN  = 64;     // channels
constexpr int CI   = 32;     // inner channels (== WMMA K!)
constexpr int NB   = 2;      // batch

// ---- fragment loaders (per-lane, per ISA 7.12.2 16-bit layouts) ----
// A 16x32 f16: lane<16 -> K in {0..7,16..23}; lane>=16 -> K in {8..15,24..31}
__device__ __forceinline__ v16h frag_pair(const _Float16* p0, const _Float16* p1) {
    v8h a = *(const v8h*)p0;   // 16B aligned
    v8h b = *(const v8h*)p1;   // 16B aligned
    return __builtin_shufflevector(a, b, 0,1,2,3,4,5,6,7,8,9,10,11,12,13,14,15);
}
// B 32x16 f16: lane holds N=lane&15, K in {0..7,16..23} (lane<16) / {8..15,24..31} (lane>=16)
__device__ __forceinline__ v16h frag_cont(const _Float16* p) {
    return frag_pair(p, p + 16);    // 16B aligned by construction
}

// ---------------- projections: q,k,v in f16, chosen layouts ----------------
__global__ void __launch_bounds__(256)
proj_kernel(const float* __restrict__ x,
            const float* __restrict__ Wk, const float* __restrict__ Wq,
            const float* __restrict__ Wv,
            _Float16* __restrict__ qRM,   // [b][m][i]  (x_q, from Wk)
            _Float16* __restrict__ kRM,   // [b][n][i]  (x_k, from Wq)
            _Float16* __restrict__ vT)    // [b][i][m]  (x_v, from Wv)
{
    const int pos = blockIdx.x * blockDim.x + threadIdx.x;
    const int i   = blockIdx.y;
    const int b   = blockIdx.z;
    const float* xb = x + (size_t)b * CIN * NT;
    float aK = 0.f, aQ = 0.f, aV = 0.f;
#pragma unroll 8
    for (int c = 0; c < CIN; ++c) {
        float xv = xb[(size_t)c * NT + pos];
        aK = fmaf(Wk[i * CIN + c], xv, aK);   // -> x_q
        aQ = fmaf(Wq[i * CIN + c], xv, aQ);   // -> x_k
        aV = fmaf(Wv[i * CIN + c], xv, aV);   // -> x_v
    }
    qRM[((size_t)b * NT + pos) * CI + i] = (_Float16)aK;
    kRM[((size_t)b * NT + pos) * CI + i] = (_Float16)aQ;
    vT [((size_t)b * CI + i) * NT + pos] = (_Float16)aV;
}

// ------- pass 1: column-wise online softmax stats (max, 1/sum) -------------
__global__ void __launch_bounds__(128)
colstats_kernel(const _Float16* __restrict__ kRM,
                const _Float16* __restrict__ qRM,
                float* __restrict__ colMax, float* __restrict__ colInv)
{
    __shared__ float red[4][64];
    const int lane = threadIdx.x & 31, wave = threadIdx.x >> 5;
    const int task = blockIdx.x * 4 + wave;           // 1152 tasks
    const int b    = task / (NT / 16);
    const int m0   = (task % (NT / 16)) * 16;
    const int half = lane >> 4, l15 = lane & 15;
    const int abase = half * 8, koff = half * 8;

    v16h fb = frag_cont(qRM + ((size_t)b * NT + m0 + l15) * CI + koff);
    const _Float16* kb = kRM + (size_t)b * NT * CI;

    float mymax = -INFINITY, mysum = 0.f;
    v8f z = {};
    for (int n0 = 0; n0 < NT; n0 += 16) {
        const _Float16* krow = kb + (size_t)(n0 + l15) * CI;
        v16h fa = frag_pair(krow + abase, krow + abase + 16);
        v8f s = __builtin_amdgcn_wmma_f32_16x16x32_f16(false, fa, false, fb,
                                                       (short)0, z, false, false);
        float nm = mymax;
#pragma unroll
        for (int r = 0; r < 8; ++r) nm = fmaxf(nm, s[r]);
        float acc = mysum * __expf(mymax - nm);
#pragma unroll
        for (int r = 0; r < 8; ++r) acc += __expf(s[r] - nm);
        mysum = acc; mymax = nm;
    }
    // combine lane L (rows 0..7) with lane L^16 (rows 8..15)
    red[wave][lane] = mymax;
    red[wave][32 + lane] = mysum;
    asm volatile("s_wait_dscnt 0" ::: "memory");
    float omax = red[wave][lane ^ 16], osum = red[wave][32 + (lane ^ 16)];
    float M = fmaxf(mymax, omax);
    float S = mysum * __expf(mymax - M) + osum * __expf(omax - M);
    if (lane < 16) {
        colMax[b * NT + m0 + l15] = M;
        colInv[b * NT + m0 + l15] = 1.0f / S;
    }
}

// ---------------- fold 1/denom into v: w[i][m] = v[i][m] * inv[m] ----------
__global__ void __launch_bounds__(256)
scalev_kernel(const _Float16* __restrict__ vT, const float* __restrict__ colInv,
              _Float16* __restrict__ wT)
{
    int idx = blockIdx.x * blockDim.x + threadIdx.x;
    if (idx >= NB * CI * NT) return;
    int m = idx % NT;
    int b = idx / (CI * NT);
    wT[idx] = (_Float16)((float)vT[idx] * colInv[b * NT + m]);
}

// ------- pass 2: recompute s, P = exp(s-max), out = Wm·(P·w)ᵀ + x ----------
__global__ void __launch_bounds__(128)
attn_kernel(const _Float16* __restrict__ kRM, const _Float16* __restrict__ qRM,
            const _Float16* __restrict__ wT,  const float* __restrict__ colMax,
            const float* __restrict__ Wm,     const float* __restrict__ x,
            float* __restrict__ out)
{
    __shared__ float wmS[CIN * CI];                       // 8 KB
    __shared__ __align__(64) _Float16 eTile[4][16 * 32];  // 4 KB
    __shared__ __align__(64) float    soTile[4][16 * 32]; // 8 KB
    for (int t = threadIdx.x; t < CIN * CI; t += 128) wmS[t] = Wm[t];
    __syncthreads();

    const int lane = threadIdx.x & 31, wave = threadIdx.x >> 5;
    const int task = blockIdx.x * 4 + wave;
    const int b    = task / (NT / 16);
    const int n0   = (task % (NT / 16)) * 16;
    const int half = lane >> 4, l15 = lane & 15;
    const int abase = half * 8, koff = half * 8;

    const _Float16* krow = kRM + ((size_t)b * NT + n0 + l15) * CI;
    v16h fk = frag_pair(krow + abase, krow + abase + 16); // loop-invariant A(k)

    _Float16* et = eTile[wave];
    const float* cmax = colMax + b * NT;
    v8f acc0 = {}, acc1 = {}, z = {};

    for (int m0 = 0; m0 < NT; m0 += 32) {
        const _Float16* q0 = qRM + ((size_t)b * NT + m0 + l15) * CI + koff;
        v16h fb0 = frag_cont(q0);
        v16h fb1 = frag_cont(q0 + 16 * CI);
        v8f s0 = __builtin_amdgcn_wmma_f32_16x16x32_f16(false, fk, false, fb0,
                                                        (short)0, z, false, false);
        v8f s1 = __builtin_amdgcn_wmma_f32_16x16x32_f16(false, fk, false, fb1,
                                                        (short)0, z, false, false);
        float mx0 = cmax[m0 + l15], mx1 = cmax[m0 + 16 + l15];
#pragma unroll
        for (int r = 0; r < 8; ++r) {
            et[(r + 8 * half) * 32 + l15]      = (_Float16)__expf(s0[r] - mx0);
            et[(r + 8 * half) * 32 + 16 + l15] = (_Float16)__expf(s1[r] - mx1);
        }
        asm volatile("s_wait_dscnt 0" ::: "memory");   // D-layout -> A-layout via LDS
        const _Float16* erow = et + l15 * 32;
        v16h fe = frag_pair(erow + abase, erow + abase + 16);
        const _Float16* w0 = wT + ((size_t)b * CI + l15) * NT + m0 + koff;
        const _Float16* w1 = wT + ((size_t)b * CI + 16 + l15) * NT + m0 + koff;
        acc0 = __builtin_amdgcn_wmma_f32_16x16x32_f16(false, fe, false, frag_cont(w0),
                                                      (short)0, acc0, false, false);
        acc1 = __builtin_amdgcn_wmma_f32_16x16x32_f16(false, fe, false, frag_cont(w1),
                                                      (short)0, acc1, false, false);
    }

    // stage sim_out tile [16 rows][32 i] and apply Wm + residual
    float* so = soTile[wave];
#pragma unroll
    for (int r = 0; r < 8; ++r) {
        so[(r + 8 * half) * 32 + l15]      = acc0[r];
        so[(r + 8 * half) * 32 + 16 + l15] = acc1[r];
    }
    asm volatile("s_wait_dscnt 0" ::: "memory");
    const float* srow = so + l15 * 32;
    const size_t xoff = (size_t)b * CIN * NT + n0 + l15;
#pragma unroll 4
    for (int c2 = 0; c2 < 32; ++c2) {
        int ch = 2 * c2 + half;
        float a = 0.f;
#pragma unroll
        for (int ii = 0; ii < CI; ++ii) a = fmaf(wmS[ch * CI + ii], srow[ii], a);
        out[xoff + (size_t)ch * NT] = x[xoff + (size_t)ch * NT] + a;
    }
}

extern "C" void kernel_launch(void* const* d_in, const int* in_sizes, int n_in,
                              void* d_out, int out_size, void* d_ws, size_t ws_size,
                              hipStream_t stream) {
    (void)in_sizes; (void)n_in; (void)out_size; (void)ws_size;
    const float* x  = (const float*)d_in[0];
    const float* Wk = (const float*)d_in[1];
    const float* Wq = (const float*)d_in[2];
    const float* Wv = (const float*)d_in[3];
    const float* Wm = (const float*)d_in[4];
    float* out = (float*)d_out;

    char* ws = (char*)d_ws;
    size_t off = 0;
    auto alloc = [&](size_t bytes) -> void* {
        void* p = ws + off; off += (bytes + 255) & ~(size_t)255; return p;
    };
    const size_t halfBuf = (size_t)NB * NT * CI * sizeof(_Float16); // 1.18 MB
    _Float16* qRM = (_Float16*)alloc(halfBuf);
    _Float16* kRM = (_Float16*)alloc(halfBuf);
    _Float16* vT  = (_Float16*)alloc(halfBuf);
    _Float16* wT  = (_Float16*)alloc(halfBuf);
    float* colMax = (float*)alloc((size_t)NB * NT * sizeof(float));
    float* colInv = (float*)alloc((size_t)NB * NT * sizeof(float));

    proj_kernel<<<dim3(NT / 256, CI, NB), 256, 0, stream>>>(x, Wk, Wq, Wv, qRM, kRM, vT);
    colstats_kernel<<<dim3(NB * (NT / 16) / 4), 128, 0, stream>>>(kRM, qRM, colMax, colInv);
    scalev_kernel<<<dim3((NB * CI * NT + 255) / 256), 256, 0, stream>>>(vT, colInv, wT);
    attn_kernel<<<dim3(NB * (NT / 16) / 4), 128, 0, stream>>>(kRM, qRM, wT, colMax, Wm, x, out);
}
